// MultiHeadAttention_53566832115758
// MI455X (gfx1250) — matrix operand order repaired and checked by reading, not compiled, not run on hardware
//
#include <hip/hip_runtime.h>

#define NTOK   131072
#define DMODEL 512
#define NCAT   1536         // q|k|v concatenated N
#define NHEAD  8
#define DHEAD  64
#define TM     64           // tokens per block
#define KC     32           // K-chunk per WMMA step
#define THREADS 512         // 16 waves (wave32) -> 4 waves/SIMD -> 256 VGPR budget
#define SEG_LD 520          // padded segment length in halfs (512 + 8)
#define ROW_LD (3 * SEG_LD) // 1560 halfs per (k-row | token-row)
#define XB_BYTES   (TM * SEG_LD * 2)         // 66560: x tile bf16
#define WBUF_BYTES (KC * ROW_LD * 2)         // 99840: one W^T chunk (q|k|v)
#define LDS_BYTES  (XB_BYTES + 2 * WBUF_BYTES) // 266240 (qkv staging reuses W buffers)

typedef __attribute__((ext_vector_type(16))) __bf16 v16bf;
typedef __attribute__((ext_vector_type(8)))  float  v8f;
typedef unsigned int v4u __attribute__((ext_vector_type(4)));
typedef int          v8i __attribute__((ext_vector_type(8)));
typedef int          v4i __attribute__((ext_vector_type(4)));

union Frag {
  uint4 q2[2];
  unsigned int u[8];
  v16bf v;
};

__device__ __forceinline__ unsigned short f2bf(float x) {
  unsigned int u = __float_as_uint(x);
  return (unsigned short)((u + 0x7FFFu + ((u >> 16) & 1u)) >> 16);  // RNE
}
__device__ __forceinline__ float bf2f(unsigned short h) {
  return __uint_as_float(((unsigned int)h) << 16);
}

// TDM: copy one 32x1536 bf16 tile (K-rows of [Wq|Wk|Wv]^T) global->LDS.
// Pad 4 DWORDs after every 256 DWORDs => 8 pad halfs after each 512-half
// segment => LDS row = [q(512)|pad|k(512)|pad|v(512)|pad] = 1560 halfs.
__device__ __forceinline__ void tdm_load_w(const unsigned short* gsrc, unsigned int lds_off) {
  unsigned long long ga = (unsigned long long)(size_t)gsrc;
  v4u g0;
  g0.x = 1u;                                                   // count=1 (valid), no gather
  g0.y = lds_off;                                              // lds_addr (bytes)
  g0.z = (unsigned int)ga;                                     // global_addr[31:0]
  g0.w = (unsigned int)((ga >> 32) & 0x01FFFFFFull) | (2u << 30); // addr[56:32] | type=2
  v8i g1;
  g1[0] = (int)((1u << 16) |                                   // data_size = 2 bytes
                (1u << 20) |                                   // pad_enable
                (7u << 22) |                                   // pad_interval: 256 DWORDs
                (3u << 25));                                   // pad_amount: 4 DWORDs
  g1[1] = (int)((unsigned)NCAT << 16);   // tensor_dim0 = 1536 (low16 at [31:16])
  g1[2] = (int)(512u << 16);             // tensor_dim0 hi=0 | tensor_dim1 low16 = 512
  g1[3] = (int)((unsigned)NCAT << 16);   // tensor_dim1 hi=0 | tile_dim0 = 1536
  g1[4] = 32;                            // tile_dim1 = 32, tile_dim2 = 0
  g1[5] = NCAT;                          // tensor_dim0_stride low32
  g1[6] = 0;
  g1[7] = 0;
  v4i z4 = {0, 0, 0, 0};
  v8i z8 = {0, 0, 0, 0, 0, 0, 0, 0};
  __builtin_amdgcn_tensor_load_to_lds(g0, g1, z4, z4, z8, 0);
}

// One-time pre-pass: Wt[k][mat*512 + j] = bf16(W_mat[j][k])  ([Wq|Wk|Wv]^T, bf16)
__global__ void pack_weights(const float* __restrict__ Wq, const float* __restrict__ Wk,
                             const float* __restrict__ Wv, unsigned short* __restrict__ Wt) {
  int idx = blockIdx.x * 256 + threadIdx.x;
  if (idx >= DMODEL * NCAT) return;
  int k   = idx / NCAT;
  int n   = idx - k * NCAT;
  int mat = n >> 9;
  int j   = n & (DMODEL - 1);
  const float* W = (mat == 0) ? Wq : (mat == 1) ? Wk : Wv;
  Wt[idx] = f2bf(W[(size_t)j * DMODEL + k]);
}

__global__ __launch_bounds__(THREADS, 1)
void fused_mha_ln(const float* __restrict__ x,
                  const unsigned short* __restrict__ Wt,   // [512][1536] bf16
                  const float* __restrict__ bq, const float* __restrict__ bk,
                  const float* __restrict__ bv,
                  const float* __restrict__ gamma, const float* __restrict__ beta,
                  float* __restrict__ out)
{
  __shared__ __align__(16) unsigned char s_raw[LDS_BYTES];
  unsigned short* sx = (unsigned short*)s_raw;                 // x tile bf16 [TM][SEG_LD]
  unsigned short* sq = (unsigned short*)(s_raw + XB_BYTES);    // qkv bf16 [TM][ROW_LD] (reuses W bufs)
  float* red = (float*)s_raw;                                  // LN partials (reuses sx)

  const int tid    = threadIdx.x;
  const int lane   = tid & 31;
  const int wave   = tid >> 5;       // 0..15
  const int nchunk = wave & 7;       // 192-column chunk of NCAT
  const int mhalf  = wave >> 3;      // rows [mhalf*32, +32)
  const int row0   = blockIdx.x * TM;
  const int half   = lane >> 4;
  const int mlo    = lane & 15;

  // ---- stage x tile (fp32 -> bf16) ----
  for (int f = tid; f < TM * DMODEL; f += THREADS) {
    int r = f >> 9, c = f & (DMODEL - 1);
    sx[r * SEG_LD + c] = f2bf(x[(size_t)(row0 + r) * DMODEL + c]);
  }

  v8f acc[2][12];
  const v8f zacc = {0.f,0.f,0.f,0.f,0.f,0.f,0.f,0.f};
  #pragma unroll
  for (int a = 0; a < 2; ++a)
    #pragma unroll
    for (int b = 0; b < 12; ++b) acc[a][b] = zacc;

  // prologue: TDM-prefetch first W chunk (wave 0 issues)
  if (wave == 0)
    tdm_load_w(Wt, (unsigned int)(size_t)(void*)(s_raw + XB_BYTES));

  // ---- GEMM: [q|k|v] = x @ Wcat, bf16 WMMA, fp32 accumulate, double-buffered TDM ----
  #pragma unroll 1   // keep rolled: full unroll spills accumulators
  for (int kci = 0; kci < DMODEL / KC; ++kci) {
    const int kc  = kci * KC;
    const int buf = kci & 1;
    const unsigned short* curw = (const unsigned short*)(s_raw + XB_BYTES + buf * WBUF_BYTES);

    __builtin_amdgcn_s_wait_tensorcnt((short)0);   // current buffer's DMA done
    __syncthreads();                               // visible to all; prev buffer reads done
    if (wave == 0 && kci + 1 < DMODEL / KC)
      tdm_load_w(Wt + (size_t)(kc + KC) * NCAT,
                 (unsigned int)(size_t)(void*)(s_raw + XB_BYTES) + (buf ^ 1) * WBUF_BYTES);

    // A fragments (16x32 bf16), ISA layout: lane = M row, K interleaved across halves
    Frag A[2];
    #pragma unroll
    for (int mt = 0; mt < 2; ++mt) {
      const unsigned short* rowp = sx + (mhalf * 32 + mt * 16 + mlo) * SEG_LD + kc;
      #pragma unroll
      for (int r = 0; r < 8; ++r) {
        int kk0 = ((r >> 2) << 4) + half * 8 + ((r & 3) << 1);
        A[mt].u[r] = *(const unsigned int*)&rowp[kk0];
      }
    }

    // B fragments (32x16 bf16): lane = N column, K interleaved across halves (same as A)
    const unsigned short* rowb = curw + mlo;
    #pragma unroll
    for (int nt = 0; nt < 12; ++nt) {
      int n0   = nchunk * 192 + nt * 16;     // tile never crosses a 512 boundary (16 | 512)
      int boff = (n0 >> 9) * SEG_LD + (n0 & (DMODEL - 1));
      Frag B;
      #pragma unroll
      for (int r = 0; r < 8; ++r) {
        int kk0 = ((r >> 2) << 4) + half * 8 + ((r & 3) << 1);
        B.u[r] = (unsigned int)rowb[kk0 * ROW_LD + boff] |
                 ((unsigned int)rowb[(kk0 + 1) * ROW_LD + boff] << 16);
      }
      acc[0][nt] = __builtin_amdgcn_wmma_f32_16x16x32_bf16(
          false, A[0].v, false, B.v, (short)0, acc[0][nt], false, false);
      acc[1][nt] = __builtin_amdgcn_wmma_f32_16x16x32_bf16(
          false, A[1].v, false, B.v, (short)0, acc[1][nt], false, false);
    }
  }
  __syncthreads();   // all W reads done: reuse region as qkv staging

  // ---- write accumulators (+bias) to LDS as bf16: sq[t][seg*SEG_LD + j] ----
  #pragma unroll
  for (int nt = 0; nt < 12; ++nt) {
    int n0  = nchunk * 192 + nt * 16;
    int seg = n0 >> 9;
    int j   = (n0 & (DMODEL - 1)) + mlo;
    const float* bias = (seg == 0) ? bq : (seg == 1) ? bk : bv;
    float bj = bias[j];
    #pragma unroll
    for (int mt = 0; mt < 2; ++mt)
      #pragma unroll
      for (int r = 0; r < 8; ++r) {
        int m = mhalf * 32 + mt * 16 + half * 8 + r;   // C/D layout: lanes 0-15 M=r, 16-31 M=8+r
        sq[m * ROW_LD + seg * SEG_LD + j] = f2bf(acc[mt][nt][r] + bj);
      }
  }
  __syncthreads();

  // ---- per-token head-vs-head attention (8x8 scores), one thread per (token, head) ----
  {
    int t = tid >> 3, h = tid & 7;                   // 512 threads == TM*NHEAD
    unsigned short* qr = sq + t * ROW_LD + 0 * SEG_LD + h * DHEAD;
    const unsigned short* kr = sq + t * ROW_LD + 1 * SEG_LD;
    const unsigned short* vr = sq + t * ROW_LD + 2 * SEG_LD;
    float s[NHEAD];
    float mx = -1e30f;
    for (int g = 0; g < NHEAD; ++g) {
      float a = 0.f;
      for (int d = 0; d < DHEAD; ++d)
        a += bf2f(qr[d]) * bf2f(kr[g * DHEAD + d]);
      s[g] = a * 0.125f;                 // 1/sqrt(64)
      mx = fmaxf(mx, s[g]);
    }
    float sum = 0.f;
    for (int g = 0; g < NHEAD; ++g) { s[g] = __expf(s[g] - mx); sum += s[g]; }
    float inv = 1.f / sum;
    for (int d = 0; d < DHEAD; ++d) {
      float c = 0.f;
      for (int g = 0; g < NHEAD; ++g) c += s[g] * bf2f(vr[g * DHEAD + d]);
      qr[d] = f2bf(c * inv);             // overwrite q segment with context
    }
  }
  __syncthreads();

  // ---- residual + LayerNorm: 8 threads per token, 64 elems each ----
  {
    int t = tid >> 3, sg = tid & 7;
    const unsigned short* cr = sq + t * ROW_LD + sg * 64;
    const float* xr = x + (size_t)(row0 + t) * DMODEL + sg * 64;
    float sum = 0.f, sq2 = 0.f;
    for (int d = 0; d < 64; ++d) {
      float y = xr[d] + bf2f(cr[d]);
      sum += y; sq2 += y * y;
    }
    red[(t * 8 + sg) * 2 + 0] = sum;
    red[(t * 8 + sg) * 2 + 1] = sq2;
  }
  __syncthreads();
  {
    int t = tid >> 3, sg = tid & 7;
    float S = 0.f, Q2 = 0.f;
    for (int i = 0; i < 8; ++i) { S += red[(t * 8 + i) * 2]; Q2 += red[(t * 8 + i) * 2 + 1]; }
    float mu   = S * (1.f / 512.f);
    float var  = Q2 * (1.f / 512.f) - mu * mu;
    float rstd = rsqrtf(var + 1e-5f);
    const unsigned short* cr = sq + t * ROW_LD + sg * 64;
    const float* xr = x + (size_t)(row0 + t) * DMODEL + sg * 64;
    float* orow = out + (size_t)(row0 + t) * DMODEL + sg * 64;
    for (int d = 0; d < 64; ++d) {
      int j = sg * 64 + d;
      float y = xr[d] + bf2f(cr[d]);
      orow[d] = (y - mu) * rstd * gamma[j] + beta[j];
    }
  }
}

extern "C" void kernel_launch(void* const* d_in, const int* in_sizes, int n_in,
                              void* d_out, int out_size, void* d_ws, size_t ws_size,
                              hipStream_t stream) {
  const float* x  = (const float*)d_in[0];
  const float* Wq = (const float*)d_in[1];
  const float* bq = (const float*)d_in[2];
  const float* Wk = (const float*)d_in[3];
  const float* bk = (const float*)d_in[4];
  const float* Wv = (const float*)d_in[5];
  const float* bv = (const float*)d_in[6];
  const float* g  = (const float*)d_in[7];
  const float* b  = (const float*)d_in[8];
  float* out = (float*)d_out;
  unsigned short* Wt = (unsigned short*)d_ws;   // 512*1536*2 = 1.5 MB

  pack_weights<<<(DMODEL * NCAT + 255) / 256, 256, 0, stream>>>(Wq, Wk, Wv, Wt);
  fused_mha_ln<<<NTOK / TM, THREADS, 0, stream>>>(x, Wt, bq, bk, bv, g, b, out);
}
